// KL_Loss_23098334117990
// MI455X (gfx1250) — matrix operand order repaired and checked by reading, not compiled, not run on hardware
//
#include <hip/hip_runtime.h>
#include <hip/hip_bf16.h>
#include <math.h>

typedef _Float16 h8   __attribute__((ext_vector_type(8)));
typedef _Float16 v16h __attribute__((ext_vector_type(16)));
typedef float    v8f  __attribute__((ext_vector_type(8)));

#define NROWS 1024
#define DDIM  128
#define K2    256
#define TEMP      0.01f
#define HALF_TEMP 0.005f
#define DIAG_VAL  (-9000000.0f * 0.01f)

// ---------------- Phase 1: build U/V (f16) and per-row sums ----------------
__global__ void kl_prep_kernel(const float* __restrict__ p1_loc,
                               const float* __restrict__ p2_loc,
                               const float* __restrict__ p1_scale,
                               const float* __restrict__ p2_scale,
                               _Float16* __restrict__ U, _Float16* __restrict__ V,
                               float* __restrict__ SLV, float* __restrict__ CT) {
    int r = blockIdx.x;            // 0..1023
    int d = threadIdx.x;           // 0..127
    const float* locp = (r < 512) ? (p1_loc   + r * DDIM) : (p2_loc   + (r - 512) * DDIM);
    const float* varp = (r < 512) ? (p1_scale + r * DDIM) : (p2_scale + (r - 512) * DDIM);
    float mu  = locp[d];
    float var = varp[d];
    float inv = 1.0f / var;
    U[r * K2 + d]        = (_Float16)(mu * mu + var);
    U[r * K2 + DDIM + d] = (_Float16)(-2.0f * mu);
    V[r * K2 + d]        = (_Float16)inv;
    V[r * K2 + DDIM + d] = (_Float16)(mu * inv);
    float lv = logf(var);
    float q  = mu * mu * inv;
    __shared__ float s_lv[128];
    __shared__ float s_q[128];
    s_lv[d] = lv; s_q[d] = q;
    __syncthreads();
    for (int off = 64; off > 0; off >>= 1) {
        if (d < off) { s_lv[d] += s_lv[d + off]; s_q[d] += s_q[d + off]; }
        __syncthreads();
    }
    if (d == 0) { SLV[r] = s_lv[0]; CT[r] = s_lv[0] + s_q[0]; }
}

// ---------------- Phase 2: exact fp32 positive-pair terms ----------------
__global__ void kl_pos_kernel(const float* __restrict__ p1_loc,
                              const float* __restrict__ p2_loc,
                              const float* __restrict__ p1_scale,
                              const float* __restrict__ p2_scale,
                              const float* __restrict__ SLV,
                              float* __restrict__ POS) {
    int i = blockIdx.x * blockDim.x + threadIdx.x;
    if (i >= NROWS) return;
    int j = (i + 512) & (NROWS - 1);
    const float* mui = (i < 512) ? (p1_loc   + i * DDIM) : (p2_loc   + (i - 512) * DDIM);
    const float* vai = (i < 512) ? (p1_scale + i * DDIM) : (p2_scale + (i - 512) * DDIM);
    const float* muj = (j < 512) ? (p1_loc   + j * DDIM) : (p2_loc   + (j - 512) * DDIM);
    const float* vaj = (j < 512) ? (p1_scale + j * DDIM) : (p2_scale + (j - 512) * DDIM);
    float s = 0.0f;
    for (int d = 0; d < DDIM; ++d) {
        float dm = mui[d] - muj[d];
        s += (dm * dm + vai[d]) / vaj[d];
    }
    POS[i] = HALF_TEMP * (SLV[j] - SLV[i] - (float)DDIM + s);
}

// ---------------- Phase 3: fused WMMA GEMM + online logsumexp ----------------
// grid = 64 blocks (16-row strips), 256 threads = 8 waves; wave w handles
// N-tiles jt = w, w+8, ..., w+56. A fragments stay resident in VGPRs.
// Per N-tile: batch-load all 8 B fragments (one clause, one wait), then issue
// 8 WMMAs back-to-back on two independent accumulator chains; prefetch the
// next tile's B rows while the matrix pipe is busy.
__global__ void __launch_bounds__(256)
kl_gemm_lse_kernel(const _Float16* __restrict__ U, const _Float16* __restrict__ V,
                   const float* __restrict__ SLV, const float* __restrict__ CT,
                   float* __restrict__ LSE) {
    const int i0   = blockIdx.x * 16;
    const int wave = threadIdx.x >> 5;
    const int lane = threadIdx.x & 31;
    const int mrow = lane & 15;   // M for A-frag, N column for B/C
    const int h    = lane >> 4;   // lane half-group

    // Resident A fragments: 8 k-tiles of 16x32 f16 (contiguous K order: K = k0 + 16h + e, same as B)
    v16h afrag[8];
    const _Float16* urow = U + (i0 + mrow) * K2;
#pragma unroll
    for (int kt = 0; kt < 8; ++kt) {
        int k0 = kt * 32;
        v16h a = *(const v16h*)(urow + k0 + 16 * h);    // K = 16h .. 16h+15
        afrag[kt] = a;
    }

    float slv_i[8];
#pragma unroll
    for (int v = 0; v < 8; ++v) slv_i[v] = SLV[i0 + v + 8 * h];

    float om[8], os[8];
#pragma unroll
    for (int v = 0; v < 8; ++v) { om[v] = -INFINITY; os[v] = 0.0f; }

    // Prefetch first tile's B rows.
    __builtin_prefetch(V + ((wave * 16) + mrow) * K2 + 16 * h, 0, 3);

    for (int t = 0; t < 8; ++t) {
        const int j0 = (wave + 8 * t) * 16;
        const int j  = j0 + mrow;
        const _Float16* vrow = V + j * K2 + 16 * h;     // B: elem e -> K = k0+16h+e

        // Batch-load all 8 B fragments (16 x b128 in one clause).
        v16h bfrag[8];
#pragma unroll
        for (int kt = 0; kt < 8; ++kt)
            bfrag[kt] = *(const v16h*)(vrow + kt * 32);

        // Prefetch next tile's B rows while WMMAs run.
        if (t < 7) {
            const _Float16* vnext = V + ((wave + 8 * (t + 1)) * 16 + mrow) * K2 + 16 * h;
            __builtin_prefetch(vnext, 0, 3);
        }

        // Two independent accumulation chains for the matrix pipe.
        v8f acc0 = {}, acc1 = {};
#pragma unroll
        for (int kt = 0; kt < 8; kt += 2) {
            acc0 = __builtin_amdgcn_wmma_f32_16x16x32_f16(
                false, afrag[kt], false, bfrag[kt], (short)0, acc0, false, false);
            acc1 = __builtin_amdgcn_wmma_f32_16x16x32_f16(
                false, afrag[kt + 1], false, bfrag[kt + 1], (short)0, acc1, false, false);
        }

        const float ctj = CT[j];
#pragma unroll
        for (int v = 0; v < 8; ++v) {
            const int i = i0 + v + 8 * h;
            float val = HALF_TEMP * (acc0[v] + acc1[v] + ctj - slv_i[v] - (float)DDIM);
            if (i == j) val = DIAG_VAL;
            float nm = fmaxf(om[v], val);
            os[v] = os[v] * expf(om[v] - nm) + expf(val - nm);
            om[v] = nm;
        }
    }

    // Reduce (max, sumexp) across waves, then across 16 column lanes.
    __shared__ float red_m[8][16][16];   // [wave][row][col]
    __shared__ float red_s[8][16][16];
#pragma unroll
    for (int v = 0; v < 8; ++v) {
        red_m[wave][v + 8 * h][mrow] = om[v];
        red_s[wave][v + 8 * h][mrow] = os[v];
    }
    __syncthreads();
    const int row = threadIdx.x >> 4;
    const int col = threadIdx.x & 15;
    float mm = -INFINITY, ss = 0.0f;
#pragma unroll
    for (int w = 0; w < 8; ++w) {
        float m2 = red_m[w][row][col];
        float s2 = red_s[w][row][col];
        float nm = fmaxf(mm, m2);
        ss = ss * expf(mm - nm) + s2 * expf(m2 - nm);
        mm = nm;
    }
    __syncthreads();
    red_m[0][row][col] = mm;
    red_s[0][row][col] = ss;
    __syncthreads();
    if (threadIdx.x < 16) {
        const int r = threadIdx.x;
        float m3 = -INFINITY, s3 = 0.0f;
        for (int c = 0; c < 16; ++c) {
            float m2 = red_m[0][r][c];
            float s2 = red_s[0][r][c];
            float nm = fmaxf(m3, m2);
            s3 = s3 * expf(m3 - nm) + s2 * expf(m2 - nm);
            m3 = nm;
        }
        LSE[i0 + r] = m3 + logf(s3);
    }
}

// ---------------- Phase 4: mean(POS - LSE) ----------------
__global__ void kl_final_kernel(const float* __restrict__ POS,
                                const float* __restrict__ LSE,
                                float* __restrict__ out) {
    __shared__ float sred[256];
    int t = threadIdx.x;
    float s = 0.0f;
    for (int i = t; i < NROWS; i += 256) s += POS[i] - LSE[i];
    sred[t] = s;
    __syncthreads();
    for (int off = 128; off > 0; off >>= 1) {
        if (t < off) sred[t] += sred[t + off];
        __syncthreads();
    }
    if (t == 0) out[0] = sred[0] * (1.0f / (float)NROWS);
}

extern "C" void kernel_launch(void* const* d_in, const int* in_sizes, int n_in,
                              void* d_out, int out_size, void* d_ws, size_t ws_size,
                              hipStream_t stream) {
    const float* p1_loc   = (const float*)d_in[0];
    const float* p2_loc   = (const float*)d_in[1];
    const float* p1_scale = (const float*)d_in[2];
    const float* p2_scale = (const float*)d_in[3];
    float* out = (float*)d_out;

    char* ws = (char*)d_ws;
    _Float16* U   = (_Float16*)(ws);                         // 1024*256*2 = 512 KB
    _Float16* V   = (_Float16*)(ws + 524288);                // 512 KB
    float*    SLV = (float*)(ws + 1048576);                  // 4 KB
    float*    CT  = (float*)(ws + 1048576 + 4096);           // 4 KB
    float*    LSE = (float*)(ws + 1048576 + 8192);           // 4 KB
    float*    POS = (float*)(ws + 1048576 + 12288);          // 4 KB

    kl_prep_kernel<<<NROWS, DDIM, 0, stream>>>(p1_loc, p2_loc, p1_scale, p2_scale,
                                               U, V, SLV, CT);
    kl_pos_kernel<<<NROWS / 256, 256, 0, stream>>>(p1_loc, p2_loc, p1_scale, p2_scale,
                                                   SLV, POS);
    kl_gemm_lse_kernel<<<NROWS / 16, 256, 0, stream>>>(U, V, SLV, CT, LSE);
    kl_final_kernel<<<1, 256, 0, stream>>>(POS, LSE, out);
}
